// CooccurrenceMatrix_36953898615251
// MI455X (gfx1250) — compile-verified, matrix operand order checked by reading, not run on hardware
//
#include <hip/hip_runtime.h>
#include <hip/hip_bf16.h>

// CooccurrenceMatrix on MI455X (gfx1250, wave32, WMMA).
//
// cooc_b = G_b * F_b^T per batch: 64 GEMMs of 512x512x400 (padded to 416).
// F entries are exact 0/1 in f16; G is split G = Ghi + Glo (f16 pair) so the
// f32-accumulating v_wmma_f32_16x16x32_f16 path reproduces f32 accuracy.
// Operands are pre-swizzled into WMMA fragment layout so the GEMM does pure
// 32B/lane global_load_b128 pairs + WMMA; the 82MB workspace is L2-resident
// (192MB L2).  Output 64MB -> HBM floor ~2.7us @ 23.3TB/s; WMMA work ~27GF.
//
// Build phase:
//   count_kernel : 1 thread / walk (32768 threads)       - counts, lens
//   frag_kernel  : 1 wave / (b,kb,tile) = 26624 waves    - fragment build
// GEMM K-loop uses three stride-advanced base pointers; paired row/col tiles
// are reached via immediate offsets (1KB apart), minimizing VALU addressing.

typedef __attribute__((ext_vector_type(16))) _Float16 v16h;
typedef __attribute__((ext_vector_type(8)))  float    v8f;

#define B_   64
#define W_   512
#define L_   20
#define V_   20
#define DV_  (L_ * V_)   // 400 true K-dim
#define KB_  13          // ceil(400/32) K-blocks of 32 (padded to 416)

// 16-bit A/B fragment K-offset for (lane, element) per CDNA5 ISA 7.12.2:
// lanes 0-15 cover K {0..7,16..23}, lanes 16-31 cover K {8..15,24..31};
// v16h elements 2j,2j+1 are the low/high halves of VGPR j.
__device__ __forceinline__ int frag_koff(int lane, int e) {
    return ((lane & 16) ? 8 : 0) + ((e & 8) ? 16 : 0) + (e & 7);
}

// Fragment storage: [b][kb][tile16][lane] -> v16h (32 bytes)
__device__ __forceinline__ size_t frag_idx(int b, int kb, int tile, int lane) {
    return ((size_t)((b * KB_ + kb) * 32 + tile) << 5) + (size_t)lane;
}

__global__ void zero_i32_kernel(int* __restrict__ p, int n) {
    int i = blockIdx.x * blockDim.x + threadIdx.x;
    if (i < n) p[i] = 0;
}

// One thread per walk: mask length, per-walk value counts, batch value counts.
__global__ __launch_bounds__(256)
void count_kernel(const int* __restrict__ nodes,
                  const int* __restrict__ masks,
                  int*   __restrict__ cnt_v,
                  int*   __restrict__ cnt_wv,
                  float* __restrict__ lens)
{
    const int i = blockIdx.x * blockDim.x + threadIdx.x;   // walk id over B_*W_
    if (i >= B_ * W_) return;
    const int b = i / W_;
    const size_t base = (size_t)i * L_;

    int nd[L_];
    unsigned mbits = 0u;
    #pragma unroll
    for (int p = 0; p < L_; ++p) {
        nd[p] = nodes[base + p];
        mbits |= (unsigned)(masks[base + p] & 1) << p;
    }

    lens[i] = (float)__popc(mbits);
    #pragma unroll
    for (int v = 0; v < V_; ++v) {
        int c = 0;
        #pragma unroll
        for (int p = 0; p < L_; ++p)
            c += (int)((mbits >> p) & 1u) & (int)(nd[p] == v);
        cnt_wv[i * V_ + v] = c;
        atomicAdd(&cnt_v[b * V_ + v], c);
    }
}

// One wave per (tile16, kb, b). Lane l handles walk tile*16 + (l%16); the two
// half-wave copies produce disjoint K offsets of the same fragment.
__global__ __launch_bounds__(32)
void frag_kernel(const int*   __restrict__ nodes,
                 const int*   __restrict__ masks,
                 const float* __restrict__ Kmat,
                 _Float16* __restrict__ F,
                 _Float16* __restrict__ Ghi,
                 _Float16* __restrict__ Glo)
{
    __shared__ float sK[L_ * V_];
    const int lane = threadIdx.x;
    const int tile = blockIdx.x;   // 0..31 (16-walk tiles)
    const int kb   = blockIdx.y;   // 0..12 (K-blocks of 32)
    const int b    = blockIdx.z;   // 0..63

    for (int i = lane; i < L_ * V_; i += 32) sK[i] = Kmat[i];
    __syncthreads();

    const int w = tile * 16 + (lane & 15);
    const size_t base = (size_t)(b * W_ + w) * L_;

    int nd[L_];
    unsigned mbits = 0u;
    #pragma unroll
    for (int p = 0; p < L_; ++p) {
        nd[p] = nodes[base + p];
        mbits |= (unsigned)(masks[base + p] & 1) << p;
    }

    v16h fv, gh, gl;
    #pragma unroll
    for (int e = 0; e < 16; ++e) {
        const int k = kb * 32 + frag_koff(lane, e);
        _Float16 fval = (_Float16)0.0f;
        float t = 0.0f;
        if (k < DV_) {
            const int q = k / V_;
            const int v = k - q * V_;
            if (((mbits >> q) & 1u) && nd[q] == v) fval = (_Float16)1.0f;
            #pragma unroll
            for (int p = 0; p < L_; ++p)
                if (((mbits >> p) & 1u) && nd[p] == v) t += sK[p * V_ + q];
        }
        fv[e] = fval;
        const _Float16 th = (_Float16)t;          // hi part
        gh[e] = th;
        gl[e] = (_Float16)(t - (float)th);        // residual -> lo part
    }
    const size_t fi = frag_idx(b, kb, tile, lane);
    ((v16h*)F)[fi]   = fv;
    ((v16h*)Ghi)[fi] = gh;
    ((v16h*)Glo)[fi] = gl;
}

__global__ void diag_kernel(const int* __restrict__ cnt_wv,
                            const int* __restrict__ cnt_v,
                            float* __restrict__ diag)
{
    int i = blockIdx.x * blockDim.x + threadIdx.x;   // over B_*W_
    if (i >= B_ * W_) return;
    const int b = i / W_;
    float d = 0.0f;
    #pragma unroll
    for (int v = 0; v < V_; ++v)
        d += (cnt_v[b * V_ + v] == 1) ? (float)cnt_wv[i * V_ + v] : 0.0f;
    diag[i] = d;
}

// 256 threads = 8 waves. Block tile: 128 rows (w) x 64 cols (x).
// Waves arranged 4x2; each wave owns a 32x32 subtile = 2x2 WMMA accumulators.
// K loop: 13 blocks of 32; per step 8 WMMAs (2x2 tiles x {hi,lo}).
__global__ __launch_bounds__(256)
void gemm_kernel(const _Float16* __restrict__ F,
                 const _Float16* __restrict__ Ghi,
                 const _Float16* __restrict__ Glo,
                 const float* __restrict__ lens,
                 const float* __restrict__ diag,
                 float* __restrict__ out)
{
    const int b    = blockIdx.z;
    const int tw   = blockIdx.y;           // 0..3 row tiles of 128
    const int tx   = blockIdx.x;           // 0..7 col tiles of 64
    const int lane = threadIdx.x & 31;
    const int wid  = threadIdx.x >> 5;     // 0..7
    const int waveRow = wid >> 1;          // 0..3
    const int waveCol = wid & 1;           // 0..1

    const int rt0 = tw * 8 + waveRow * 2;  // 16-row tile index (0..31)
    const int ct0 = tx * 4 + waveCol * 2;  // 16-col tile index (0..31)

    // Stride-advanced fragment pointers; kb stride = 32 tiles * 32 lanes v16h.
    const size_t KSTEP = 32 * 32;                       // v16h per K-block
    const v16h* pAh = (const v16h*)Ghi + frag_idx(b, 0, rt0, lane);
    const v16h* pAl = (const v16h*)Glo + frag_idx(b, 0, rt0, lane);
    const v16h* pB  = (const v16h*)F   + frag_idx(b, 0, ct0, lane);

    v8f acc00 = {}, acc01 = {}, acc10 = {}, acc11 = {};

    for (int kb = 0; kb < KB_; ++kb) {
        const v16h a0h = pAh[0];           // rt0   tile, hi
        const v16h a1h = pAh[32];          // rt0+1 tile, hi   (+1KB imm offset)
        const v16h a0l = pAl[0];           // rt0   tile, lo
        const v16h a1l = pAl[32];          // rt0+1 tile, lo
        const v16h b0  = pB[0];            // ct0   tile
        const v16h b1  = pB[32];           // ct0+1 tile
        pAh += KSTEP; pAl += KSTEP; pB += KSTEP;

        acc00 = __builtin_amdgcn_wmma_f32_16x16x32_f16(false, a0h, false, b0, (short)0, acc00, false, false);
        acc00 = __builtin_amdgcn_wmma_f32_16x16x32_f16(false, a0l, false, b0, (short)0, acc00, false, false);
        acc01 = __builtin_amdgcn_wmma_f32_16x16x32_f16(false, a0h, false, b1, (short)0, acc01, false, false);
        acc01 = __builtin_amdgcn_wmma_f32_16x16x32_f16(false, a0l, false, b1, (short)0, acc01, false, false);
        acc10 = __builtin_amdgcn_wmma_f32_16x16x32_f16(false, a1h, false, b0, (short)0, acc10, false, false);
        acc10 = __builtin_amdgcn_wmma_f32_16x16x32_f16(false, a1l, false, b0, (short)0, acc10, false, false);
        acc11 = __builtin_amdgcn_wmma_f32_16x16x32_f16(false, a1h, false, b1, (short)0, acc11, false, false);
        acc11 = __builtin_amdgcn_wmma_f32_16x16x32_f16(false, a1l, false, b1, (short)0, acc11, false, false);
    }

    // Epilogue: C/D layout is VGPR r -> M = r + 8*(lane>=16), N = lane%16.
    const int n    = lane & 15;
    const int mofs = (lane >> 4) * 8;
    const float* lensB = lens + b * W_;
    const float* diagB = diag + b * W_;
    float* outB = out + (size_t)b * W_ * W_;

    const v8f accs[4] = {acc00, acc01, acc10, acc11};
    #pragma unroll
    for (int t = 0; t < 4; ++t) {
        const int i = t >> 1, j = t & 1;
        const int rowBase = (rt0 + i) * 16;
        const int gx = (ct0 + j) * 16 + n;
        const float lx = lensB[gx];
        #pragma unroll
        for (int r = 0; r < 8; ++r) {
            const int gw = rowBase + mofs + r;
            float val = accs[t][r];
            if (gw == gx) val -= diagB[gw];
            val /= (lensB[gw] * lx + 1e-8f);
            outB[(size_t)gw * W_ + gx] = val;
        }
    }
}

extern "C" void kernel_launch(void* const* d_in, const int* in_sizes, int n_in,
                              void* d_out, int out_size, void* d_ws, size_t ws_size,
                              hipStream_t stream)
{
    const int*   nodes = (const int*)d_in[0];    // anonymized_nodes (B,W,L)
    const int*   masks = (const int*)d_in[1];    // walk_masks       (B,W,L)
    const float* Kmat  = (const float*)d_in[2];  // kernel           (20,20)

    char* ws = (char*)d_ws;
    const size_t fragBytes = (size_t)B_ * KB_ * 32 * 32 * 16 * 2;  // 27,262,976 B each

    _Float16* F    = (_Float16*)(ws);
    _Float16* Ghi  = (_Float16*)(ws + fragBytes);
    _Float16* Glo  = (_Float16*)(ws + 2 * fragBytes);
    size_t off     = 3 * fragBytes;
    int*   cnt_v   = (int*)(ws + off);            off += (size_t)B_ * V_ * 4;
    int*   cnt_wv  = (int*)(ws + off);            off += (size_t)B_ * W_ * V_ * 4;
    float* diag    = (float*)(ws + off);          off += (size_t)B_ * W_ * 4;
    float* lens    = (float*)(ws + off);          // total ~84.7 MB

    zero_i32_kernel<<<(B_ * V_ + 255) / 256, 256, 0, stream>>>(cnt_v, B_ * V_);
    count_kernel<<<(B_ * W_ + 255) / 256, 256, 0, stream>>>(nodes, masks,
                                                            cnt_v, cnt_wv, lens);
    frag_kernel<<<dim3(32, KB_, B_), 32, 0, stream>>>(nodes, masks, Kmat, F, Ghi, Glo);
    diag_kernel<<<(B_ * W_ + 255) / 256, 256, 0, stream>>>(cnt_wv, cnt_v, diag);
    gemm_kernel<<<dim3(8, 4, B_), 256, 0, stream>>>(F, Ghi, Glo, lens, diag, (float*)d_out);
}
